// ErwinTransolver_65274912964977
// MI455X (gfx1250) — compile-verified, matrix operand order checked by reading, not run on hardware
//
#include <hip/hip_runtime.h>
#include <hip/hip_bf16.h>

// ---------------------------------------------------------------------------
// MI455X (gfx1250) implementation of the Transolver+Erwin reference.
// Big GEMMs -> v_wmma_f32_16x16x32_bf16 with fragment-linear LDS tiles
// (A row-major [m][k], B transposed [n][k], 16B-aligned rows so each WMMA
// fragment is two ds_load_b128). float4 global staging on the coalesced axis.
// Small/irregular ops -> wave32 VALU kernels.
// ---------------------------------------------------------------------------

#define BATCH 4
#define NTOK  8192
#define NHEAD 8
#define DHEAD 64
#define NSLC  64
#define BHROW (BATCH*NHEAD)            // 32
#define NROWS ((size_t)BHROW*NTOK)     // 262144 (b,h,n) rows
#define KSPLIT 8                       // split-K factor for k_eid

typedef __bf16 bf16_t;
typedef __attribute__((ext_vector_type(16))) __bf16 v16bf;
typedef __attribute__((ext_vector_type(8)))  __bf16 v8bf;
typedef __attribute__((ext_vector_type(4)))  __bf16 v4bf;
typedef __attribute__((ext_vector_type(8)))  float  v8f;
typedef __attribute__((ext_vector_type(4)))  float  v4f;

static_assert(sizeof(bf16_t) == 2, "bf16 size");

// ---- helpers ---------------------------------------------------------------

__device__ __forceinline__ bf16_t f2bf(float f) {
  union { float f; unsigned u; } v; v.f = f;
  unsigned r = v.u + 0x7fffu + ((v.u >> 16) & 1u);   // round to nearest even
  unsigned short h = (unsigned short)(r >> 16);
  return __builtin_bit_cast(bf16_t, h);
}

__device__ __forceinline__ v4bf f2bf4(v4f x) {
  v4bf o;
  o[0] = f2bf(x[0]); o[1] = f2bf(x[1]); o[2] = f2bf(x[2]); o[3] = f2bf(x[3]);
  return o;
}

__device__ __forceinline__ float geluf(float x) {  // tanh approximation (jax default)
  const float c0 = 0.7978845608028654f, c1 = 0.044715f;
  return 0.5f * x * (1.0f + tanhf(c0 * (x + c1 * x * x * x)));
}

__device__ __forceinline__ float wred_sum(float v) {
  #pragma unroll
  for (int m = 16; m; m >>= 1) v += __shfl_xor(v, m, 32);
  return v;
}
__device__ __forceinline__ float wred_max(float v) {
  #pragma unroll
  for (int m = 16; m; m >>= 1) v = fmaxf(v, __shfl_xor(v, m, 32));
  return v;
}

// ---- WMMA fragment machinery (ISA 7.12.2 layouts, wave32) ------------------
// Tiles are stored fragment-linear: operand row r (= m for A, = n for B^T),
// 32 K-values per row, row stride LDT=40 bf16 (80B, 16B aligned). Per-lane
// fragment = two contiguous 16B chunks: K in [half*8, half*8+8) and
// [16+half*8, 16+half*8+8)  (half = lane>>4, row = lane&15).

#define LDT 40

__device__ __forceinline__ v16bf load_frag(const bf16_t* T) {
  int lane = threadIdx.x & 31;
  int row = lane & 15, half = (lane >> 4) & 1;
  const bf16_t* rp = T + row * LDT + half * 8;
  v8bf lo = *(const v8bf*)rp;          // ds_load_b128
  v8bf hi = *(const v8bf*)(rp + 16);   // ds_load_b128
  return __builtin_shufflevector(lo, hi, 0, 1, 2, 3, 4, 5, 6, 7,
                                 8, 9, 10, 11, 12, 13, 14, 15);
}

__device__ __forceinline__ v8f wmma_bf16(v16bf a, v16bf b, v8f c) {
  return __builtin_amdgcn_wmma_f32_16x16x32_bf16(false, a, false, b, (short)0, c,
                                                 false, false);
}

// 4 waves (2x2) per 128-thread WG cover a 64x64 output tile; one K=32 step.
__device__ __forceinline__ void wave_mma(const bf16_t* As, const bf16_t* Bt,
                                         int wr, int wc, v8f (&acc)[2][2]) {
  v16bf a0 = load_frag(As + (wr * 32) * LDT);
  v16bf a1 = load_frag(As + (wr * 32 + 16) * LDT);
  v16bf b0 = load_frag(Bt + (wc * 32) * LDT);
  v16bf b1 = load_frag(Bt + (wc * 32 + 16) * LDT);
  acc[0][0] = wmma_bf16(a0, b0, acc[0][0]);
  acc[0][1] = wmma_bf16(a0, b1, acc[0][1]);
  acc[1][0] = wmma_bf16(a1, b0, acc[1][0]);
  acc[1][1] = wmma_bf16(a1, b1, acc[1][1]);
}

// Staging helpers (128 threads):
// contiguous 64x32 A tile from row-major src (ld floats): float4 -> v4bf store
__device__ __forceinline__ void stage_a_rows(bf16_t* As, const float* src,
                                             size_t ld, int tid) {
  #pragma unroll
  for (int it = 0; it < 4; ++it) {
    int u = tid + it * 128;            // 512 float4 units
    int r = u >> 3, c4 = (u & 7) << 2;
    v4f x = *(const v4f*)(src + (size_t)r * ld + c4);
    *(v4bf*)(As + r * LDT + c4) = f2bf4(x);
  }
}
// transposed 32x64 -> Bt[n][k]: read float4 along n (coalesced), scatter bf16
__device__ __forceinline__ void stage_b_trans(bf16_t* Bt, const float* src,
                                              size_t ld, int tid) {
  #pragma unroll
  for (int it = 0; it < 4; ++it) {
    int u = tid + it * 128;            // 512 float4 units
    int k = u >> 4, n4 = (u & 15) << 2;
    v4f x = *(const v4f*)(src + (size_t)k * ld + n4);
    #pragma unroll
    for (int j = 0; j < 4; ++j) Bt[(n4 + j) * LDT + k] = f2bf(x[j]);
  }
}

// ---- Phase A: xp = (x @ Wx + bx) scattered to (B,H,N,DH) -------------------

__global__ __launch_bounds__(128) void k_xp(const float* __restrict__ X,
                                            const float* __restrict__ Wx,
                                            const float* __restrict__ bx,
                                            float* __restrict__ xp) {
  __shared__ __align__(16) bf16_t As[64 * LDT];
  __shared__ __align__(16) bf16_t Bt[64 * LDT];
  const int tid = threadIdx.x, lane = tid & 31, wave = tid >> 5;
  const int wr = wave >> 1, wc = wave & 1;
  const int n0 = blockIdx.x * 64, m0 = blockIdx.y * 64;
  v8f acc[2][2] = {};
  for (int k0 = 0; k0 < 256; k0 += 32) {
    stage_a_rows(As, X + (size_t)m0 * 256 + k0, 256, tid);
    stage_b_trans(Bt, Wx + (size_t)k0 * 512 + n0, 512, tid);
    __syncthreads();
    wave_mma(As, Bt, wr, wc, acc);
    __syncthreads();
  }
  int ln = lane & 15, hs = lane >> 4;
  #pragma unroll
  for (int mi = 0; mi < 2; ++mi)
    #pragma unroll
    for (int ni = 0; ni < 2; ++ni)
      #pragma unroll
      for (int v = 0; v < 8; ++v) {
        int m = m0 + wr * 32 + mi * 16 + v + 8 * hs;
        int col = n0 + wc * 32 + ni * 16 + ln;
        int b = m >> 13, n = m & 8191, h = col >> 6, dd = col & 63;
        xp[(((size_t)b * NHEAD + h) * NTOK + n) * 64 + dd] =
            acc[mi][ni][v] + bx[col];
      }
}

// ---- logits = xp @ Ws + bs (row-major (B,H,N,G)) ---------------------------

__global__ __launch_bounds__(128) void k_logits(const float* __restrict__ xp,
                                                const float* __restrict__ Ws,
                                                const float* __restrict__ bs,
                                                float* __restrict__ lg) {
  __shared__ __align__(16) bf16_t As[64 * LDT];
  __shared__ __align__(16) bf16_t Bt[64 * LDT];
  const int tid = threadIdx.x, lane = tid & 31, wave = tid >> 5;
  const int wr = wave >> 1, wc = wave & 1;
  const int m0 = blockIdx.y * 64;
  v8f acc[2][2] = {};
  for (int k0 = 0; k0 < 64; k0 += 32) {
    stage_a_rows(As, xp + (size_t)m0 * 64 + k0, 64, tid);
    stage_b_trans(Bt, Ws + (size_t)k0 * 64, 64, tid);
    __syncthreads();
    wave_mma(As, Bt, wr, wc, acc);
    __syncthreads();
  }
  int ln = lane & 15, hs = lane >> 4;
  #pragma unroll
  for (int mi = 0; mi < 2; ++mi)
    #pragma unroll
    for (int ni = 0; ni < 2; ++ni)
      #pragma unroll
      for (int v = 0; v < 8; ++v) {
        int m = m0 + wr * 32 + mi * 16 + v + 8 * hs;
        int col = wc * 32 + ni * 16 + ln;
        lg[(size_t)m * 64 + col] = acc[mi][ni][v] + bs[col];
      }
}

// ---- pp = (gelu(pos@Wp1+bp1)@Wp2+bp2) to (B,H,N,3) -------------------------

__global__ void k_pp(const float* __restrict__ pos, const float* __restrict__ Wp1,
                     const float* __restrict__ bp1, const float* __restrict__ Wp2,
                     const float* __restrict__ bp2, float* __restrict__ pp) {
  int id = blockIdx.x * blockDim.x + threadIdx.x;
  if (id >= BATCH * NTOK) return;
  int b = id / NTOK, n = id % NTOK;
  float p0 = pos[(size_t)id * 3], p1 = pos[(size_t)id * 3 + 1],
        p2 = pos[(size_t)id * 3 + 2];
  float t1[24];
  #pragma unroll
  for (int j = 0; j < 24; ++j)
    t1[j] = geluf(bp1[j] + p0 * Wp1[j] + p1 * Wp1[24 + j] + p2 * Wp1[48 + j]);
  #pragma unroll
  for (int k2 = 0; k2 < 24; ++k2) {
    float a = bp2[k2];
    #pragma unroll
    for (int j = 0; j < 24; ++j) a += t1[j] * Wp2[j * 24 + k2];
    int h = k2 / 3, y = k2 % 3;
    pp[(((size_t)b * NHEAD + h) * NTOK + n) * 3 + y] = a;
  }
}

// ---- fused LN -> tau -> softmax (sw in place of logits), wave per row ------

__global__ void k_softmax(const float* __restrict__ xp, float* __restrict__ sw,
                          const float* __restrict__ ln_g,
                          const float* __restrict__ ln_b,
                          const float* __restrict__ Wt,
                          const float* __restrict__ bt) {
  size_t row = (size_t)blockIdx.x * 8 + (threadIdx.x >> 5);
  if (row >= NROWS) return;
  int lane = threadIdx.x & 31;
  const float* xr = xp + row * 64;
  float x0 = xr[lane], x1 = xr[lane + 32];
  float mean = wred_sum(x0 + x1) * (1.0f / 64.0f);
  float d0 = x0 - mean, d1 = x1 - mean;
  float var = wred_sum(d0 * d0 + d1 * d1) * (1.0f / 64.0f);
  float inv = rsqrtf(var + 1e-5f);
  float n0 = d0 * inv * ln_g[lane] + ln_b[lane];
  float n1 = d1 * inv * ln_g[lane + 32] + ln_b[lane + 32];
  float t = wred_sum(n0 * Wt[lane] + n1 * Wt[lane + 32]);
  float tau = fminf(2.0f, fmaxf(0.1f, 0.5f + t + bt[0]));
  float* lr = sw + row * 64;
  float rinv_tau = 1.0f / tau;
  float l0 = lr[lane] * rinv_tau, l1 = lr[lane + 32] * rinv_tau;
  float mx = wred_max(fmaxf(l0, l1));
  float e0 = __expf(l0 - mx), e1 = __expf(l1 - mx);
  float den = 1.0f / wred_sum(e0 + e1);
  lr[lane] = e0 * den;
  lr[lane + 32] = e1 * den;
}

// ---- norm[b,h,g] = sum_n sw ------------------------------------------------

__global__ void k_norm(const float* __restrict__ sw, float* __restrict__ norm) {
  __shared__ float red[256];
  int bh = blockIdx.x >> 6, g = blockIdx.x & 63;
  float s = 0.f;
  for (int n = threadIdx.x; n < NTOK; n += 256)
    s += sw[((size_t)bh * NTOK + n) * 64 + g];
  red[threadIdx.x] = s;
  __syncthreads();
  for (int st = 128; st; st >>= 1) {
    if ((int)threadIdx.x < st) red[threadIdx.x] += red[threadIdx.x + st];
    __syncthreads();
  }
  if (threadIdx.x == 0) norm[blockIdx.x] = red[0];
}

// ---- eid partials: per (bh, chunk) 64x64 GEMM over 1024 K ------------------
// A = sw^T (transpose staged in LDS), B = xp. 256 WGs for full-chip streaming.

__global__ __launch_bounds__(128) void k_eid_part(const float* __restrict__ sw,
                                                  const float* __restrict__ xp,
                                                  float* __restrict__ part) {
  __shared__ __align__(16) bf16_t As[64 * LDT];
  __shared__ __align__(16) bf16_t Bt[64 * LDT];
  const int tid = threadIdx.x, lane = tid & 31, wave = tid >> 5;
  const int wr = wave >> 1, wc = wave & 1;
  const int bh = blockIdx.x, chunk = blockIdx.y;
  const size_t base = (size_t)bh * NTOK;
  const int kbeg = chunk * (NTOK / KSPLIT), kend = kbeg + NTOK / KSPLIT;
  v8f acc[2][2] = {};
  for (int k0 = kbeg; k0 < kend; k0 += 32) {
    // A[g][kk] = sw[(base+k0+kk)*64+g]: read coalesced along g, scatter in LDS
    #pragma unroll
    for (int it = 0; it < 4; ++it) {
      int u = tid + it * 128;
      int kk = u >> 4, g4 = (u & 15) << 2;
      v4f x = *(const v4f*)(sw + (base + k0 + kk) * 64 + g4);
      #pragma unroll
      for (int j = 0; j < 4; ++j) As[(g4 + j) * LDT + kk] = f2bf(x[j]);
    }
    // B^T[d][kk] = xp[(base+k0+kk)*64+d]: read coalesced along d, scatter
    #pragma unroll
    for (int it = 0; it < 4; ++it) {
      int u = tid + it * 128;
      int kk = u >> 4, d4 = (u & 15) << 2;
      v4f x = *(const v4f*)(xp + (base + k0 + kk) * 64 + d4);
      #pragma unroll
      for (int j = 0; j < 4; ++j) Bt[(d4 + j) * LDT + kk] = f2bf(x[j]);
    }
    __syncthreads();
    wave_mma(As, Bt, wr, wc, acc);
    __syncthreads();
  }
  float* po = part + ((size_t)chunk * BHROW + bh) * 4096;
  int ln = lane & 15, hs = lane >> 4;
  #pragma unroll
  for (int mi = 0; mi < 2; ++mi)
    #pragma unroll
    for (int ni = 0; ni < 2; ++ni)
      #pragma unroll
      for (int v = 0; v < 8; ++v) {
        int g = wr * 32 + mi * 16 + v + 8 * hs;
        int dd = wc * 32 + ni * 16 + ln;
        po[g * 64 + dd] = acc[mi][ni][v];
      }
}

__global__ void k_eid_red(const float* __restrict__ part,
                          const float* __restrict__ norm,
                          float* __restrict__ eid) {
  int id = blockIdx.x * blockDim.x + threadIdx.x;  // 32*4096
  if (id >= BHROW * 4096) return;
  int bh = id >> 12, rem = id & 4095, g = rem >> 6;
  float s = 0.f;
  #pragma unroll
  for (int c = 0; c < KSPLIT; ++c)
    s += part[((size_t)c * BHROW + bh) * 4096 + rem];
  eid[id] = s / (norm[bh * 64 + g] + 1e-3f);
}

// ---- eip[b,h,g,y] -----------------------------------------------------------

__global__ void k_eip(const float* __restrict__ sw, const float* __restrict__ pp,
                      const float* __restrict__ norm, float* __restrict__ eip) {
  int bh = blockIdx.x, tid = threadIdx.x;
  if (tid >= 192) return;
  int g = tid / 3, y = tid % 3;
  float s = 0.f;
  for (int n = 0; n < NTOK; ++n)
    s += sw[((size_t)bh * NTOK + n) * 64 + g] * pp[((size_t)bh * NTOK + n) * 3 + y];
  eip[(bh * 64 + g) * 3 + y] = s / (norm[bh * 64 + g] + 1e-3f);
}

// ---- generic small VALU GEMM (embed / pool / unpool) -----------------------

__global__ void k_gemm_small(const float* __restrict__ A, int lda,
                             const float* __restrict__ W,
                             const float* __restrict__ bias,
                             const float* __restrict__ resid,
                             float* __restrict__ C, int M, int N, int K) {
  int id = blockIdx.x * blockDim.x + threadIdx.x;
  if (id >= M * N) return;
  int m = id / N, n = id % N;
  float a = bias ? bias[n] : 0.f;
  const float* ar = A + (size_t)m * lda;
  for (int k = 0; k < K; ++k) a += ar[k] * W[(size_t)k * N + n];
  if (resid) a += resid[id];
  C[id] = a;
}

__global__ void k_pos2(const float* __restrict__ eip, float* __restrict__ pos2) {
  int id = blockIdx.x * blockDim.x + threadIdx.x;
  if (id >= 1024 * 3) return;
  int i = id / 3, y = id % 3;
  pos2[id] = 0.5f * (eip[(2 * i) * 3 + y] + eip[(2 * i + 1) * 3 + y]);
}

// ---- Erwin ball-attention block: one workgroup per ball, all in LDS --------

struct BlkP {
  const float *n1_g, *n1_b, *qkv_W, *qkv_b, *proj_W, *proj_b, *pe_W, *pe_b,
      *sigma, *n2_g, *n2_b, *m1_W, *m1_b, *m2_W, *m2_b;
};

__device__ __forceinline__ void ln_rows(const float* src, float* dst,
                                        const float* g, const float* b, int ball,
                                        int d, int tid) {
  if (tid < ball) {
    const float* xr = src + tid * d;
    float m = 0.f;
    for (int c = 0; c < d; ++c) m += xr[c];
    m /= (float)d;
    float v = 0.f;
    for (int c = 0; c < d; ++c) { float t = xr[c] - m; v += t * t; }
    v /= (float)d;
    float inv = rsqrtf(v + 1e-5f);
    float* dr = dst + tid * d;
    for (int c = 0; c < d; ++c) dr[c] = (xr[c] - m) * inv * g[c] + b[c];
  }
}

__global__ __launch_bounds__(128) void k_block(float* __restrict__ x,
                                               const float* __restrict__ pos,
                                               BlkP p, int d, int ball,
                                               int group_len, int shift) {
  const int T = 128, tid = threadIdx.x;
  const int heads = 8, dh = d / heads;
  __shared__ float s_x[2048];   // ball*d (residual stream)
  __shared__ float s_n[2048];   // LN output / attn input
  __shared__ float s_o[2048];   // attention output (pre-proj)
  __shared__ float s_h[8192];   // qkv (3*ball*d) then MLP hidden (ball*4d)
  __shared__ float s_sc[1024];  // scores (ball*ball)
  __shared__ float s_p[32 * 3];
  __shared__ float s_pm[3];

  int bpg = group_len / ball;
  int grp = blockIdx.x / bpg, kb = blockIdx.x % bpg;
  int sh = shift ? (ball >> 1) : 0;
  int base = grp * group_len;
  auto gidx = [&](int t) { return base + ((kb * ball + t + sh) % group_len); };

  int bd = ball * d;
  for (int i = tid; i < bd; i += T) {
    int t = i / d, c = i % d;
    s_x[i] = x[(size_t)gidx(t) * d + c];
  }
  for (int i = tid; i < ball * 3; i += T) {
    int t = i / 3, c = i % 3;
    s_p[i] = pos[(size_t)gidx(t) * 3 + c];
  }
  __syncthreads();

  ln_rows(s_x, s_n, p.n1_g, p.n1_b, ball, d, tid);
  if (tid < 3) {
    float m = 0.f;
    for (int t = 0; t < ball; ++t) m += s_p[t * 3 + tid];
    s_pm[tid] = m / (float)ball;
  }
  __syncthreads();

  // relative positional encoding added to LN output
  for (int i = tid; i < bd; i += T) {
    int t = i / d, c = i % d;
    float r0 = s_p[t * 3] - s_pm[0], r1 = s_p[t * 3 + 1] - s_pm[1],
          r2 = s_p[t * 3 + 2] - s_pm[2];
    s_n[i] += r0 * p.pe_W[c] + r1 * p.pe_W[d + c] + r2 * p.pe_W[2 * d + c] +
              p.pe_b[c];
  }
  __syncthreads();

  // qkv projection: s_h[comp*bd + t*d + (head*dh+i)]
  int d3 = 3 * d;
  for (int i = tid; i < ball * d3; i += T) {
    int t = i / d3, j = i % d3;
    float a = p.qkv_b[j];
    const float* xr = s_n + t * d;
    for (int c = 0; c < d; ++c) a += xr[c] * p.qkv_W[(size_t)c * d3 + j];
    int comp = j / d, rem = j - comp * d;
    s_h[comp * bd + t * d + rem] = a;
  }
  __syncthreads();

  const float* q = s_h;
  const float* kk = s_h + bd;
  const float* vv = s_h + 2 * bd;
  float scl = rsqrtf((float)dh);
  for (int h = 0; h < heads; ++h) {
    float sg = p.sigma[h];
    for (int i = tid; i < ball * ball; i += T) {
      int a = i / ball, b = i % ball;
      float s = 0.f;
      const float* qa = q + a * d + h * dh;
      const float* kb2 = kk + b * d + h * dh;
      for (int ii = 0; ii < dh; ++ii) s += qa[ii] * kb2[ii];
      float dx = s_p[a * 3] - s_p[b * 3], dy = s_p[a * 3 + 1] - s_p[b * 3 + 1],
            dz = s_p[a * 3 + 2] - s_p[b * 3 + 2];
      s_sc[i] = s * scl + sg * sqrtf(dx * dx + dy * dy + dz * dz);
    }
    __syncthreads();
    if (tid < ball) {
      float mx = -1e30f;
      for (int j = 0; j < ball; ++j) mx = fmaxf(mx, s_sc[tid * ball + j]);
      float se = 0.f;
      for (int j = 0; j < ball; ++j) {
        float e = __expf(s_sc[tid * ball + j] - mx);
        s_sc[tid * ball + j] = e;
        se += e;
      }
      float r = 1.0f / se;
      for (int j = 0; j < ball; ++j) s_sc[tid * ball + j] *= r;
    }
    __syncthreads();
    for (int i = tid; i < ball * dh; i += T) {
      int t = i / dh, ii = i % dh;
      float o = 0.f;
      for (int j = 0; j < ball; ++j) o += s_sc[t * ball + j] * vv[j * d + h * dh + ii];
      s_o[t * d + h * dh + ii] = o;
    }
    __syncthreads();
  }

  // output projection + residual
  for (int i = tid; i < bd; i += T) {
    int t = i / d, c = i % d;
    float a = p.proj_b[c];
    const float* orow = s_o + t * d;
    for (int j = 0; j < d; ++j) a += orow[j] * p.proj_W[(size_t)j * d + c];
    s_x[i] += a;
  }
  __syncthreads();

  ln_rows(s_x, s_n, p.n2_g, p.n2_b, ball, d, tid);
  __syncthreads();

  // MLP: gelu(h @ m1 + m1_b) @ m2 + m2_b, residual
  int d4 = 4 * d;
  for (int i = tid; i < ball * d4; i += T) {
    int t = i / d4, j = i % d4;
    float a = p.m1_b[j];
    const float* xr = s_n + t * d;
    for (int c = 0; c < d; ++c) a += xr[c] * p.m1_W[(size_t)c * d4 + j];
    s_h[i] = geluf(a);
  }
  __syncthreads();
  for (int i = tid; i < bd; i += T) {
    int t = i / d, c = i % d;
    float a = p.m2_b[c];
    const float* hr = s_h + t * d4;
    for (int j = 0; j < d4; ++j) a += hr[j] * p.m2_W[(size_t)j * d + c];
    s_x[i] += a;
  }
  __syncthreads();

  for (int i = tid; i < bd; i += T) {
    int t = i / d, c = i % d;
    x[(size_t)gidx(t) * d + c] = s_x[i];
  }
}

// ---- M_buf[b,h,g,:] = upd[b,h,g,:] @ Wo[h*64: h*64+64, :] ------------------

__global__ void k_updwo(const float* __restrict__ upd, const float* __restrict__ Wo,
                        float* __restrict__ Mb) {
  int id = blockIdx.x * blockDim.x + threadIdx.x;  // 2048*256
  int r = id >> 8, c = id & 255;
  int h = (r >> 6) & 7;
  float a = 0.f;
  const float* ur = upd + (size_t)r * 64;
  for (int dd = 0; dd < 64; ++dd) a += ur[dd] * Wo[(size_t)(h * 64 + dd) * 256 + c];
  Mb[id] = a;
}

// ---- fused final: y[b,n,c] = sum_{h,g} sw * Mb + bo  (WMMA, K=512) ---------

__global__ __launch_bounds__(128) void k_final(const float* __restrict__ sw,
                                               const float* __restrict__ Mb,
                                               const float* __restrict__ bo,
                                               float* __restrict__ out) {
  __shared__ __align__(16) bf16_t As[64 * LDT];
  __shared__ __align__(16) bf16_t Bt[64 * LDT];
  const int tid = threadIdx.x, lane = tid & 31, wave = tid >> 5;
  const int wr = wave >> 1, wc = wave & 1;
  const int n0 = blockIdx.x * 64, m0 = blockIdx.y * 64, b = blockIdx.z;
  v8f acc[2][2] = {};
  for (int k0 = 0; k0 < 512; k0 += 32) {
    int h = k0 >> 6, g0 = k0 & 63;
    size_t swb = ((size_t)(b * NHEAD + h)) * NTOK;
    stage_a_rows(As, sw + (swb + m0) * 64 + g0, 64, tid);
    stage_b_trans(Bt, Mb + ((size_t)b * 512 + k0) * 256 + n0, 256, tid);
    __syncthreads();
    wave_mma(As, Bt, wr, wc, acc);
    __syncthreads();
  }
  int ln = lane & 15, hs = lane >> 4;
  #pragma unroll
  for (int mi = 0; mi < 2; ++mi)
    #pragma unroll
    for (int ni = 0; ni < 2; ++ni)
      #pragma unroll
      for (int v = 0; v < 8; ++v) {
        int m = m0 + wr * 32 + mi * 16 + v + 8 * hs;
        int col = n0 + wc * 32 + ni * 16 + ln;
        out[((size_t)b * NTOK + m) * 256 + col] = acc[mi][ni][v] + bo[col];
      }
}

// ---------------------------------------------------------------------------
// Host side
// ---------------------------------------------------------------------------

// depth-first insertion-order flattening of setup_inputs()
enum {
  IX_X = 0, IX_POS = 1, IX_WX = 2, IX_BX = 3, IX_WP1 = 4, IX_BP1 = 5,
  IX_WP2 = 6, IX_BP2 = 7, IX_LNG = 8, IX_LNB = 9, IX_WT = 10, IX_BT = 11,
  IX_WS = 12, IX_BS = 13, IX_WO = 14, IX_BO = 15, IX_EMBW = 16, IX_EMBB = 17,
  IX_ENC0 = 18,          // 4 blocks x 15 leaves
  IX_POOLW = 78, IX_POOLB = 79,
  IX_BOT0 = 80,          // 4 blocks x 15 leaves
  IX_UNPW = 140, IX_UNPB = 141,
  IX_DEC0 = 142          // 4 blocks x 15 leaves
};

static BlkP mk_blk(void* const* d_in, int b0) {
  BlkP p;
  p.n1_g   = (const float*)d_in[b0 + 0];
  p.n1_b   = (const float*)d_in[b0 + 1];
  p.qkv_W  = (const float*)d_in[b0 + 2];
  p.qkv_b  = (const float*)d_in[b0 + 3];
  p.proj_W = (const float*)d_in[b0 + 4];
  p.proj_b = (const float*)d_in[b0 + 5];
  p.pe_W   = (const float*)d_in[b0 + 6];
  p.pe_b   = (const float*)d_in[b0 + 7];
  p.sigma  = (const float*)d_in[b0 + 8];
  p.n2_g   = (const float*)d_in[b0 + 9];
  p.n2_b   = (const float*)d_in[b0 + 10];
  p.m1_W   = (const float*)d_in[b0 + 11];
  p.m1_b   = (const float*)d_in[b0 + 12];
  p.m2_W   = (const float*)d_in[b0 + 13];
  p.m2_b   = (const float*)d_in[b0 + 14];
  return p;
}

extern "C" void kernel_launch(void* const* d_in, const int* in_sizes, int n_in,
                              void* d_out, int out_size, void* d_ws,
                              size_t ws_size, hipStream_t stream) {
  (void)in_sizes; (void)n_in; (void)out_size; (void)ws_size;

  const float* X   = (const float*)d_in[IX_X];
  const float* POS = (const float*)d_in[IX_POS];
  float* OUT = (float*)d_out;

  // workspace carve-up (floats, 256-aligned)
  float* w = (float*)d_ws;
  size_t off = 0;
  auto alloc = [&](size_t n) {
    float* p = w + off;
    off += (n + 255) & ~(size_t)255;
    return p;
  };
  float* xp   = alloc((size_t)BHROW * NTOK * 64);  // 16.7M
  float* sw   = alloc((size_t)BHROW * NTOK * 64);  // logits -> softmax weights
  float* pp   = alloc((size_t)BHROW * NTOK * 3);
  float* nrm  = alloc(BHROW * 64);
  float* part = alloc((size_t)KSPLIT * BHROW * 4096);  // eid split-K partials
  float* eid  = alloc(2048 * 64);
  float* eip  = alloc(2048 * 3);
  float* xcur = alloc(2048 * 64);
  float* skip = alloc(2048 * 64);
  float* x2   = alloc(1024 * 128);
  float* pos2 = alloc(1024 * 3);
  float* Mb   = alloc(2048 * 256);

  // Phase A: projections
  k_xp<<<dim3(8, 512), 128, 0, stream>>>(X, (const float*)d_in[IX_WX],
                                         (const float*)d_in[IX_BX], xp);
  k_pp<<<(BATCH * NTOK + 255) / 256, 256, 0, stream>>>(
      POS, (const float*)d_in[IX_WP1], (const float*)d_in[IX_BP1],
      (const float*)d_in[IX_WP2], (const float*)d_in[IX_BP2], pp);

  // Phase B: slice weights
  k_logits<<<dim3(1, 4096), 128, 0, stream>>>(xp, (const float*)d_in[IX_WS],
                                              (const float*)d_in[IX_BS], sw);
  k_softmax<<<32768, 256, 0, stream>>>(xp, sw, (const float*)d_in[IX_LNG],
                                       (const float*)d_in[IX_LNB],
                                       (const float*)d_in[IX_WT],
                                       (const float*)d_in[IX_BT]);
  k_norm<<<BHROW * 64, 256, 0, stream>>>(sw, nrm);
  k_eid_part<<<dim3(BHROW, KSPLIT), 128, 0, stream>>>(sw, xp, part);
  k_eid_red<<<(BHROW * 4096 + 255) / 256, 256, 0, stream>>>(part, nrm, eid);
  k_eip<<<BHROW, 192, 0, stream>>>(sw, pp, nrm, eip);

  // Phase C: Erwin U-net on 2048 tokens of dim 64
  k_gemm_small<<<(2048 * 64 + 255) / 256, 256, 0, stream>>>(
      eid, 64, (const float*)d_in[IX_EMBW], (const float*)d_in[IX_EMBB],
      nullptr, xcur, 2048, 64, 64);
  for (int i = 0; i < 4; ++i) {
    BlkP p = mk_blk(d_in, IX_ENC0 + 15 * i);
    k_block<<<64, 128, 0, stream>>>(xcur, eip, p, 64, 32, 64, i & 1);
  }
  hipMemcpyAsync(skip, xcur, 2048 * 64 * sizeof(float),
                 hipMemcpyDeviceToDevice, stream);
  k_gemm_small<<<(1024 * 128 + 255) / 256, 256, 0, stream>>>(
      xcur, 128, (const float*)d_in[IX_POOLW], (const float*)d_in[IX_POOLB],
      nullptr, x2, 1024, 128, 128);
  k_pos2<<<(1024 * 3 + 255) / 256, 256, 0, stream>>>(eip, pos2);
  for (int i = 0; i < 4; ++i) {
    BlkP p = mk_blk(d_in, IX_BOT0 + 15 * i);
    k_block<<<64, 128, 0, stream>>>(x2, pos2, p, 128, 16, 32, i & 1);
  }
  k_gemm_small<<<(1024 * 128 + 255) / 256, 256, 0, stream>>>(
      x2, 128, (const float*)d_in[IX_UNPW], (const float*)d_in[IX_UNPB],
      skip, xcur, 1024, 128, 128);
  for (int i = 0; i < 4; ++i) {
    BlkP p = mk_blk(d_in, IX_DEC0 + 15 * i);
    k_block<<<64, 128, 0, stream>>>(xcur, eip, p, 64, 32, 64, i & 1);
  }

  // Phase D: fused de-slice + output projection
  k_updwo<<<2048, 256, 0, stream>>>(xcur, (const float*)d_in[IX_WO], Mb);
  k_final<<<dim3(4, 128, 4), 128, 0, stream>>>(sw, Mb,
                                               (const float*)d_in[IX_BO], OUT);
}
